// SpatialBranch_vsgnet_67774583931069
// MI455X (gfx1250) — matrix operand order repaired and checked by reading, not compiled, not run on hardware
//
#include <hip/hip_runtime.h>
#include <stdint.h>

typedef __attribute__((ext_vector_type(16))) _Float16 v16h;
typedef __attribute__((ext_vector_type(8)))  _Float16 v8h;
typedef __attribute__((ext_vector_type(8)))  float    v8f;
typedef __attribute__((ext_vector_type(4)))  int      v4i;

#define C1K   50        // 2*5*5
#define C1KP  64        // padded K for conv1
#define C2K   1600      // 64*5*5
#define C1OUT 64
#define C2OUT 32
#define MIDW  30        // valid pooled conv1 width/height
#define MIDP  36        // padded row width; cols 30-31 absorb pool garbage, 32-35 stay zero
#define MIDS  (MIDW*MIDP)   // 1080 halves per channel
#define LDIM  512

// ---- dynamic LDS carve (255,360 B < 320 KB WGP LDS) ----
#define MID_BYTES  (C1OUT*MIDS*2)          // 138240
#define W1_OFF     MID_BYTES
#define W1_BYTES   (C1OUT*C1KP*2)          //   8192
#define W2_OFF     (W1_OFF + W1_BYTES)
#define W2_BYTES   (C2OUT*C2K*2)           // 102400
#define ACC_OFF    (W2_OFF + W2_BYTES)
#define ACC_BYTES  (C2OUT*4)
#define OFFS_OFF   (ACC_OFF + ACC_BYTES)
#define OFFS_BYTES (C2K*4)                 //   6400
#define SMEM_TOTAL (OFFS_OFF + OFFS_BYTES) // 255360

// compare-select max (values never NaN on kept paths; avoids v_max canonicalize trio)
static __device__ __forceinline__ float selmax(float a, float b) { return (a > b) ? a : b; }

// 2x2 maxpool butterflies in DPP (no LDS round-trip)
static __device__ __forceinline__ float dpp_xor1_max(float x) {
    int i = __builtin_bit_cast(int, x);
    int p = __builtin_amdgcn_update_dpp(i, i, 0xB1 /*quad_perm(1,0,3,2)*/, 0xF, 0xF, true);
    return selmax(x, __builtin_bit_cast(float, p));
}
static __device__ __forceinline__ float dpp_xor8_max(float x) {
    int i = __builtin_bit_cast(int, x);
    int p = __builtin_amdgcn_update_dpp(i, i, 0x128 /*row_ror:8*/, 0xF, 0xF, true);
    return selmax(x, __builtin_bit_cast(float, p));
}

__global__ __launch_bounds__(256)
void vsgnet_spatial_fused(const float* __restrict__ bboxes,
                          const int*   __restrict__ obj_pairs,
                          const int*   __restrict__ frame_w,
                          const int*   __restrict__ frame_h,
                          const float* __restrict__ W1,
                          const float* __restrict__ b1,
                          const float* __restrict__ W2,
                          const float* __restrict__ b2,
                          const float* __restrict__ Wl,
                          const float* __restrict__ bl,
                          float*       __restrict__ out)
{
    extern __shared__ char smem[];
    _Float16* mid  = (_Float16*)(smem);
    _Float16* w1h  = (_Float16*)(smem + W1_OFF);
    _Float16* w2h  = (_Float16*)(smem + W2_OFF);
    float*    macc = (float*)  (smem + ACC_OFF);
    int*      offs = (int*)    (smem + OFFS_OFF);

    const int tid  = threadIdx.x;
    const int lane = tid & 31;
    const int wave = tid >> 5;
    const int t    = blockIdx.x;      // relation pair 0..1023
    const int bimg = t >> 6;
    const int rrel = t & 63;

    // ---------------- phase 0: staging ----------------
    const float sfx = 64.0f / (float)frame_w[0];
    const float sfy = 64.0f / (float)frame_h[0];
    int bx0[2], bx1[2], by0[2], by1[2];
    #pragma unroll
    for (int s = 0; s < 2; ++s) {
        const int obj = obj_pairs[(bimg*64 + rrel)*2 + s];
        const float* bb = bboxes + (size_t)(bimg*36 + obj)*4;
        bx0[s] = (int)floorf(bb[0]*sfx);
        by0[s] = (int)floorf(bb[1]*sfy);
        bx1[s] = (int)floorf(bb[2]*sfx);
        by1[s] = (int)floorf(bb[3]*sfy);
    }
    {   // zero mid: keeps never-written halo cols 32-35 finite-zero for conv2 gathers
        uint32_t* mz = (uint32_t*)mid;
        for (int i = tid; i < (C1OUT*MIDS)/2; i += 256) mz[i] = 0u;
    }
    for (int i = tid; i < C1OUT*C1KP; i += 256) {        // W1 -> f16, K pad 50->64
        const int mm = i >> 6, kk = i & 63;
        w1h[i] = (kk < C1K) ? (_Float16)W1[mm*C1K + kk] : (_Float16)0.f;
    }
    for (int i = tid; i < C2OUT*C2K; i += 256)           // W2 -> f16
        w2h[i] = (_Float16)W2[i];
    for (int k = tid; k < C2K; k += 256) {               // conv2 im2col LUT in BYTE units
        const int c = k / 25, r = k % 25;
        offs[k] = (c*MIDS + (r/5)*MIDP + (r%5)) * 2;
    }
    if (tid < C2OUT) macc[tid] = 0.f;
    __builtin_prefetch(Wl + tid, 0, 1);                  // global_prefetch_b8
    __syncthreads();

    const int m  = lane & 15;
    const int n  = lane & 15;
    const int hi = lane >> 4;
    const bool rep = ((lane & 1) == 0) && ((lane & 8) == 0);

    // two ds_load_b128 per A fragment (halves 0-7 at K0, 8-15 at K0+8)
    auto ldA = [&](const _Float16* base, int off) -> v16h {
        v8h lo = *(const v8h*)(base + off);
        v8h hp = *(const v8h*)(base + off + 8);
        return __builtin_shufflevector(lo, hp, 0,1,2,3,4,5,6,7,8,9,10,11,12,13,14,15);
    };

    // bias fragments: C-layout row m = e + 8*hi -> bias is a v8f slice
    v8f bias1[4], bias2[2];
    #pragma unroll
    for (int mt = 0; mt < 4; ++mt) bias1[mt] = *(const v8f*)(b1 + mt*16 + hi*8);
    #pragma unroll
    for (int mt = 0; mt < 2; ++mt) bias2[mt] = *(const v8f*)(b2 + mt*16 + hi*8);

    // ---------------- phase 1: conv1 implicit GEMM [64 x 50p64] x [50 x 3600] + pool ----------------
    v16h a1[4][2];
    #pragma unroll
    for (int mt = 0; mt < 4; ++mt)
        #pragma unroll
        for (int ks = 0; ks < 2; ++ks)
            a1[mt][ks] = ldA(w1h, (mt*16 + m)*C1KP + ks*32 + hi*16);

    for (int nt = wave; nt < 240; nt += 8) {
        const int by  = nt >> 3, bxg = nt & 7;
        const int oy  = by*2  + (n >> 3);
        const int ox  = bxg*8 + (n & 7);

        // per-map 25-bit window mask: bit(ky*5+kx) = pixel(oy+ky, ox+kx) inside box
        unsigned w25[2];
        #pragma unroll
        for (int s = 0; s < 2; ++s) {
            unsigned rb = 0, cb = 0;
            #pragma unroll
            for (int j = 0; j < 5; ++j) {
                rb |= (unsigned)((oy + j >= by0[s]) & (oy + j < by1[s])) << j;
                cb |= (unsigned)((ox + j >= bx0[s]) & (ox + j < bx1[s])) << j;
            }
            unsigned w = 0;
            #pragma unroll
            for (int ky = 0; ky < 5; ++ky)
                w |= ((rb >> ky) & 1u) ? (cb << (5*ky)) : 0u;
            w25[s] = w;
        }
        // funnel fields: halves h of each k-step are contiguous bit-ranges
        unsigned flds[2];
        flds[0] = hi ? (((w25[0] >> 16) & 0x1FFu) | (w25[1] << 9)) : w25[0];
        flds[1] = hi ? ((w25[1] >> 23) & 0x3u)                     : (w25[1] >> 7);

        v8f acc[4] = { bias1[0], bias1[1], bias1[2], bias1[3] };
        #pragma unroll
        for (int ks = 0; ks < 2; ++ks) {
            const unsigned f = flds[ks];
            v16h bf;
            #pragma unroll
            for (int h = 0; h < 16; ++h)
                bf[h] = __builtin_bit_cast(_Float16,
                          (unsigned short)(((f >> h) & 1u) ? 0x3C00u : 0u));
            #pragma unroll
            for (int mt = 0; mt < 4; ++mt)
                acc[mt] = __builtin_amdgcn_wmma_f32_16x16x32_f16(
                    false, a1[mt][ks], false, bf, (short)0, acc[mt], false, false);
        }

        // unconditional epilogue: pool partners hold identical values (benign same-addr
        // stores); px 30/31 garbage lands in the padded halo columns conv2 never keeps.
        const int px  = bxg*4 + ((n & 7) >> 1);
        const int rowb = by*MIDP + px;
        #pragma unroll
        for (int mt = 0; mt < 4; ++mt)
            #pragma unroll
            for (int e = 0; e < 8; ++e) {
                const float x = dpp_xor8_max(dpp_xor1_max(acc[mt][e]));
                const int c = mt*16 + e + hi*8;
                mid[c*MIDS + rowb] = (_Float16)x;
            }
    }
    __syncthreads();

    // ---------------- phase 2: conv2 implicit GEMM [32 x 1600] x [1600 x 676] + pool + mean ----------------
    const char* midb = (const char*)mid;
    v8f psum[2] = {};                          // per-thread pooled-sum accumulators
    for (int nt0 = wave; nt0 < 52; nt0 += 16) {
        const int ntA = nt0, ntB = nt0 + 8;    // ntB may be >= 52 (computed, masked out)
        const int basenA = (((ntA >> 2)*2 + (n >> 3))*MIDP + (ntA & 3)*8 + (n & 7)) * 2;
        const int basenB = (((ntB >> 2)*2 + (n >> 3))*MIDP + (ntB & 3)*8 + (n & 7)) * 2;
        v8f accA[2] = { bias2[0], bias2[1] };
        v8f accB[2] = { bias2[0], bias2[1] };
        for (int ks = 0; ks < 50; ++ks) {      // K = 1600, 50 exact steps of 32
            const int k0 = ks*32 + hi*16;
            v16h bfA, bfB;
            #pragma unroll
            for (int g = 0; g < 4; ++g) {
                const v4i o = *(const v4i*)(offs + k0 + g*4);   // ds_load_b128 (byte offs)
                #pragma unroll
                for (int j = 0; j < 4; ++j) {
                    bfA[g*4 + j] = *(const _Float16*)(midb + o[j] + basenA);
                    bfB[g*4 + j] = *(const _Float16*)(midb + o[j] + basenB);
                }
            }
            #pragma unroll
            for (int mt = 0; mt < 2; ++mt) {
                const v16h af = ldA(w2h, (mt*16 + m)*C2K + ks*32 + hi*16);
                accA[mt] = __builtin_amdgcn_wmma_f32_16x16x32_f16(
                    false, af, false, bfA, (short)0, accA[mt], false, false);
                accB[mt] = __builtin_amdgcn_wmma_f32_16x16x32_f16(
                    false, af, false, bfB, (short)0, accB[mt], false, false);
            }
        }
        // branchless: accumulate valid pooled values into registers
        #pragma unroll
        for (int sel = 0; sel < 2; ++sel) {
            const int nt = sel ? ntB : ntA;
            const int px = (nt & 3)*4 + ((n & 7) >> 1);
            const bool valid = (nt < 52) && (px < 13);
            #pragma unroll
            for (int mt = 0; mt < 2; ++mt)
                #pragma unroll
                for (int e = 0; e < 8; ++e) {
                    const float x = dpp_xor8_max(dpp_xor1_max(sel ? accB[mt][e] : accA[mt][e]));
                    psum[mt][e] += valid ? x : 0.f;
                }
        }
    }
    if (rep) {   // one guarded block: 16 ds_add_f32 per representative lane
        #pragma unroll
        for (int mt = 0; mt < 2; ++mt)
            #pragma unroll
            for (int e = 0; e < 8; ++e)
                atomicAdd(&macc[mt*16 + e + hi*8], psum[mt][e]);
    }
    __syncthreads();

    // ---------------- phase 3: mean (/169) -> linear 32->512 -> ReLU ----------------
    for (int j = tid; j < LDIM; j += 256) {
        float s = bl[j];
        #pragma unroll
        for (int c = 0; c < C2OUT; ++c)
            s += (macc[c] * (1.f/169.f)) * Wl[c*LDIM + j];
        out[(size_t)t*LDIM + j] = fmaxf(s, 0.f);
    }
}

extern "C" void kernel_launch(void* const* d_in, const int* in_sizes, int n_in,
                              void* d_out, int out_size, void* d_ws, size_t ws_size,
                              hipStream_t stream) {
    (void)in_sizes; (void)n_in; (void)d_ws; (void)ws_size; (void)out_size;
    const float* bboxes    = (const float*)d_in[0];
    const int*   obj_pairs = (const int*)  d_in[1];
    /* d_in[2] = num_rels (always R in reference; unused) */
    const int*   fw        = (const int*)  d_in[3];
    const int*   fh        = (const int*)  d_in[4];
    const float* W1        = (const float*)d_in[5];
    const float* b1        = (const float*)d_in[6];
    const float* W2        = (const float*)d_in[7];
    const float* b2        = (const float*)d_in[8];
    const float* Wl        = (const float*)d_in[9];
    const float* bl        = (const float*)d_in[10];
    float* out = (float*)d_out;

    static_assert(SMEM_TOTAL <= 320*1024, "fits CDNA5 WGP LDS");
    (void)hipFuncSetAttribute(reinterpret_cast<const void*>(&vsgnet_spatial_fused),
                              hipFuncAttributeMaxDynamicSharedMemorySize, SMEM_TOTAL);

    vsgnet_spatial_fused<<<dim3(1024), dim3(256), SMEM_TOTAL, stream>>>(
        bboxes, obj_pairs, fw, fh, W1, b1, W2, b2, Wl, bl, out);
}
